// ModalityAlignmentModel_2_55997783605469
// MI455X (gfx1250) — matrix operand order repaired and checked by reading, not compiled, not run on hardware
//
#include <hip/hip_runtime.h>
#include <math.h>

// ---------------------------------------------------------------------------
// Problem geometry
// ---------------------------------------------------------------------------
#define CONV_H   628
#define CONV_W   512
#define SPATIAL  (CONV_H * CONV_W)      // 321536
#define AUDIO_K  SPATIAL                // 321536
#define AUDIO_N  1024
#define TEXT_M   8192
#define TEXT_K   3072
#define TEXT_N   256
#define N_SUB    4
#define SUB_D    256
#define TOPK     5

#define KSTEP    32                     // bf16 WMMA K per instruction
#define AUDIO_KSTEPS (AUDIO_K / KSTEP)  // 10048
#define SPLITK   64
#define KSTEPS_PER_SPLIT (AUDIO_KSTEPS / SPLITK) // 157

// output layout: [audio 16x1024][text 8192x256][loss]
#define OUT_AUDIO_OFF 0
#define OUT_TEXT_OFF  (16 * AUDIO_N)                     // 16384
#define OUT_LOSS_OFF  (OUT_TEXT_OFF + TEXT_M * TEXT_N)   // 2113536

// workspace layout (floats)
#define WS_H_OFF   0
#define WS_SC_OFF  (16 * SPATIAL)       // scores/probs 4*16*8192

typedef __attribute__((ext_vector_type(16))) __bf16 v16bf;
typedef __attribute__((ext_vector_type(8)))  float  v8f;

// ---------------------------------------------------------------------------
// fragment builders: 16 consecutive-K fp32 values -> bf16 fragment lanes
// ---------------------------------------------------------------------------
__device__ __forceinline__ v16bf cvt16(float4 p0, float4 p1, float4 p2, float4 p3) {
  v16bf r;
  r[0]  = (__bf16)p0.x; r[1]  = (__bf16)p0.y; r[2]  = (__bf16)p0.z; r[3]  = (__bf16)p0.w;
  r[4]  = (__bf16)p1.x; r[5]  = (__bf16)p1.y; r[6]  = (__bf16)p1.z; r[7]  = (__bf16)p1.w;
  r[8]  = (__bf16)p2.x; r[9]  = (__bf16)p2.y; r[10] = (__bf16)p2.z; r[11] = (__bf16)p2.w;
  r[12] = (__bf16)p3.x; r[13] = (__bf16)p3.y; r[14] = (__bf16)p3.z; r[15] = (__bf16)p3.w;
  return r;
}

// A fragment (16x32 bf16): lane (hf,m): contiguous run [kk+hf*16 .. +15]
__device__ __forceinline__ v16bf load_a_frag(const float* __restrict__ arow, int kk, int hf) {
  const float* p = arow + kk + hf * 16;
  float4 a0 = *(const float4*)(p);
  float4 a1 = *(const float4*)(p + 4);
  float4 a2 = *(const float4*)(p + 8);
  float4 a3 = *(const float4*)(p + 12);
  return cvt16(a0, a1, a2, a3);
}

// B fragment (32x16 bf16): lane (hf,n): contiguous run [kk+hf*16 .. +15] along B-source row n
__device__ __forceinline__ v16bf load_b_frag(const float* __restrict__ brow, int kk, int hf) {
  const float* p = brow + kk + hf * 16;
  float4 b0 = *(const float4*)(p);
  float4 b1 = *(const float4*)(p + 4);
  float4 b2 = *(const float4*)(p + 8);
  float4 b3 = *(const float4*)(p + 12);
  return cvt16(b0, b1, b2, b3);
}

// async staging: copy one 64B per-lane run of wa into LDS (ASYNCcnt-tracked)
__device__ __forceinline__ void issue_chunk(const float* __restrict__ base,
                                            unsigned gbyte, unsigned lbyte) {
  asm volatile("global_load_async_to_lds_b128 %0, %1, %2"
               :: "v"(lbyte),       "v"(gbyte),       "s"(base) : "memory");
  asm volatile("global_load_async_to_lds_b128 %0, %1, %2"
               :: "v"(lbyte + 16u), "v"(gbyte + 16u), "s"(base) : "memory");
  asm volatile("global_load_async_to_lds_b128 %0, %1, %2"
               :: "v"(lbyte + 32u), "v"(gbyte + 32u), "s"(base) : "memory");
  asm volatile("global_load_async_to_lds_b128 %0, %1, %2"
               :: "v"(lbyte + 48u), "v"(gbyte + 48u), "s"(base) : "memory");
}

__device__ __forceinline__ float softplus_f(float x) {
  return fmaxf(x, 0.0f) + log1pf(__expf(-fabsf(x)));
}

// ---------------------------------------------------------------------------
// 0) init: audio bias into output accumulator, zero the loss slot
// ---------------------------------------------------------------------------
__global__ void init_out_kernel(const float* __restrict__ ba, float* __restrict__ out) {
  int i = blockIdx.x * blockDim.x + threadIdx.x;
  if (i < 16 * AUDIO_N) out[OUT_AUDIO_OFF + i] = ba[i & (AUDIO_N - 1)];
  if (i == 16 * AUDIO_N) out[OUT_LOSS_OFF] = 0.0f;
}

// ---------------------------------------------------------------------------
// 1) conv 1->16, 3x3, pad 1 + ReLU ; h[ch][p] row-major (16 x 321536)
// ---------------------------------------------------------------------------
__global__ void conv_relu_kernel(const float* __restrict__ x,
                                 const float* __restrict__ w,
                                 const float* __restrict__ b,
                                 float* __restrict__ h) {
  int p = blockIdx.x * blockDim.x + threadIdx.x;
  if (p >= SPATIAL) return;
  int r = p >> 9, c = p & 511;
  float n[9];
#pragma unroll
  for (int dy = 0; dy < 3; ++dy)
#pragma unroll
    for (int dx = 0; dx < 3; ++dx) {
      int rr = r + dy - 1, cc = c + dx - 1;
      n[dy * 3 + dx] = (rr >= 0 && rr < CONV_H && cc >= 0 && cc < CONV_W)
                           ? x[rr * CONV_W + cc] : 0.0f;
    }
  for (int ch = 0; ch < 16; ++ch) {
    float acc = b[ch];
#pragma unroll
    for (int i = 0; i < 9; ++i) acc += n[i] * w[ch * 9 + i];
    h[ch * SPATIAL + p] = fmaxf(acc, 0.0f);
  }
}

// ---------------------------------------------------------------------------
// 2) audio GEMM: (16 x 321536) x (321536 x 1024), split-K + f32 atomics.
//    HBM-bound on wa (1.317 GB). The wa stream runs through a 4-deep
//    ASYNCcnt LDS pipeline (global_load_async_to_lds_b128): zero VGPR cost
//    for in-flight data, consume via ds_load_b128 after s_wait_asynccnt.
// ---------------------------------------------------------------------------
__global__ void audio_gemm_kernel(const float* __restrict__ h,
                                  const float* __restrict__ wa,
                                  float* __restrict__ out) {
  __shared__ float bbuf[4][16 * KSTEP];   // 4 x 2KB staging for wa chunks

  int tile  = blockIdx.x & 63;   // N-tile (64 tiles of 16)
  int split = blockIdx.x >> 6;   // K split id
  int lane  = threadIdx.x;       // wave32
  int m  = lane & 15;
  int hf = lane >> 4;
  int n0 = tile * 16;

  const float* arow = h + (size_t)m * AUDIO_K;           // A row M=m (L2-resident)
  // per-lane byte offset of its 16-float run in wa (fits u32: wa < 1.32 GB)
  unsigned grow = ((unsigned)(n0 + m) * (unsigned)AUDIO_K + (unsigned)(hf * 16)) * 4u;
  unsigned lrow = (unsigned)(m * KSTEP + hf * 16) * 4u;  // byte offset inside a buffer
  unsigned lds0 = (unsigned)(unsigned long long)(void*)&bbuf[0][0];

  int kk0 = split * KSTEPS_PER_SPLIT * KSTEP;

  // prologue: fill 3 of the 4 buffers
#pragma unroll
  for (int p = 0; p < 3; ++p)
    issue_chunk(wa, grow + (unsigned)((kk0 + p * KSTEP) * 4), lds0 + (unsigned)(p * 2048) + lrow);

  v8f acc = {};
  for (int s = 0; s < KSTEPS_PER_SPLIT; ++s) {
    int kk = kk0 + s * KSTEP;
    if (s + 3 < KSTEPS_PER_SPLIT) {
      // buffer (s+3)&3 was consumed at step s-1: ensure its ds reads retired
      asm volatile("s_wait_dscnt 0x0" ::: "memory");
      issue_chunk(wa, grow + (unsigned)((kk + 3 * KSTEP) * 4),
                  lds0 + (unsigned)(((s + 3) & 3) * 2048) + lrow);
      asm volatile("s_wait_asynccnt 0xc" ::: "memory");  // chunk s complete, 3 in flight
    } else {
      asm volatile("s_wait_asynccnt 0x0" ::: "memory");  // drain tail
    }

    __builtin_prefetch(arow + kk + 8 * KSTEP, 0, 3);     // stream-ahead on h
    v16bf A = load_a_frag(arow, kk, hf);

    // B from LDS staging buffer (each lane reads back its staged 16 floats)
    const float* bp = &bbuf[s & 3][m * KSTEP + hf * 16];
    float4 b0 = *(const float4*)(bp);
    float4 b1 = *(const float4*)(bp + 4);
    float4 b2 = *(const float4*)(bp + 8);
    float4 b3 = *(const float4*)(bp + 12);
    v16bf B = cvt16(b0, b1, b2, b3);

    acc = __builtin_amdgcn_wmma_f32_16x16x32_bf16(false, A, false, B,
                                                  (short)0, acc, false, false);
  }

  // C layout: VGPR r -> M = r + 8*hf, N = n0 + (lane&15)
  int cn = n0 + (lane & 15);
#pragma unroll
  for (int r = 0; r < 8; ++r)
    atomicAdd(&out[OUT_AUDIO_OFF + (hf * 8 + r) * AUDIO_N + cn], acc[r]);
}

// ---------------------------------------------------------------------------
// 3) text GEMM: (8192 x 3072) x (3072 x 256), one wave per 16x16 tile
// ---------------------------------------------------------------------------
__global__ void text_gemm_kernel(const float* __restrict__ tf,
                                 const float* __restrict__ wt,
                                 const float* __restrict__ bt,
                                 float* __restrict__ out) {
  int nt   = blockIdx.x & 15;    // 16 N-tiles
  int mt   = blockIdx.x >> 4;    // 512 M-tiles
  int lane = threadIdx.x;
  int m  = lane & 15;
  int hf = lane >> 4;

  const float* arow = tf + (size_t)(mt * 16 + m) * TEXT_K;
  const float* brow = wt + (size_t)(nt * 16 + m) * TEXT_K;

  v8f acc = {};
  for (int kk = 0; kk < TEXT_K; kk += KSTEP) {
    v16bf A = load_a_frag(arow, kk, hf);
    v16bf B = load_b_frag(brow, kk, hf);
    acc = __builtin_amdgcn_wmma_f32_16x16x32_bf16(false, A, false, B,
                                                  (short)0, acc, false, false);
  }
  int cn = nt * 16 + (lane & 15);
  float bias = bt[cn];
#pragma unroll
  for (int r = 0; r < 8; ++r)
    out[OUT_TEXT_OFF + (mt * 16 + hf * 8 + r) * TEXT_N + cn] = acc[r] + bias;
}

// ---------------------------------------------------------------------------
// 4) scores: for each sub g: (16 x 256) x (256 x 8192) -> ws scores
// ---------------------------------------------------------------------------
__global__ void scores_gemm_kernel(const float* __restrict__ out,  // reads audio+text
                                   float* __restrict__ sc) {
  int nt   = blockIdx.x & 511;   // 512 N-tiles over 8192
  int g    = blockIdx.x >> 9;    // sub-embedding group
  int lane = threadIdx.x;
  int m  = lane & 15;
  int hf = lane >> 4;

  const float* arow = out + OUT_AUDIO_OFF + (size_t)m * AUDIO_N + g * SUB_D;
  const float* brow = out + OUT_TEXT_OFF + (size_t)(nt * 16 + m) * TEXT_N;

  v8f acc = {};
#pragma unroll
  for (int kk = 0; kk < SUB_D; kk += KSTEP) {
    v16bf A = load_a_frag(arow, kk, hf);
    v16bf B = load_b_frag(brow, kk, hf);
    acc = __builtin_amdgcn_wmma_f32_16x16x32_bf16(false, A, false, B,
                                                  (short)0, acc, false, false);
  }
  int cn = nt * 16 + (lane & 15);
#pragma unroll
  for (int r = 0; r < 8; ++r)
    sc[(size_t)g * 16 * TEXT_M + (hf * 8 + r) * TEXT_M + cn] = acc[r];
}

// ---------------------------------------------------------------------------
// 5) softmax over N=8192, one block per (g,b) row, in place in ws
// ---------------------------------------------------------------------------
__global__ void softmax_kernel(float* __restrict__ sc) {
  __shared__ float red[256];
  float* row = sc + (size_t)blockIdx.x * TEXT_M;
  int tid = threadIdx.x;

  float mx = -1e30f;
  for (int i = tid; i < TEXT_M; i += 256) mx = fmaxf(mx, row[i]);
  red[tid] = mx; __syncthreads();
  for (int s = 128; s > 0; s >>= 1) {
    if (tid < s) red[tid] = fmaxf(red[tid], red[tid + s]);
    __syncthreads();
  }
  mx = red[0]; __syncthreads();

  float sum = 0.0f;
  for (int i = tid; i < TEXT_M; i += 256) {
    float e = __expf(row[i] - mx);
    row[i] = e;
    sum += e;
  }
  red[tid] = sum; __syncthreads();
  for (int s = 128; s > 0; s >>= 1) {
    if (tid < s) red[tid] += red[tid + s];
    __syncthreads();
  }
  float inv = 1.0f / red[0];
  for (int i = tid; i < TEXT_M; i += 256) row[i] *= inv;
}

// ---------------------------------------------------------------------------
// 6) per-group top-5 / bottom-5 over 131072 probs + BCE-softplus loss
// ---------------------------------------------------------------------------
__global__ void topk_loss_kernel(const float* __restrict__ probs,
                                 float* __restrict__ out) {
  __shared__ float st[256 * TOPK];
  __shared__ float sb[256 * TOPK];
  int g = blockIdx.x, tid = threadIdx.x;
  const float* p = probs + (size_t)g * 16 * TEXT_M;
  const int n = 16 * TEXT_M;

  float top[TOPK], bot[TOPK];
#pragma unroll
  for (int i = 0; i < TOPK; ++i) { top[i] = -1e30f; bot[i] = 1e30f; }
  for (int i = tid; i < n; i += 256) {
    float v = p[i];
    if (v > top[TOPK - 1]) {
      int j = TOPK - 1;
      while (j > 0 && top[j - 1] < v) { top[j] = top[j - 1]; --j; }
      top[j] = v;
    }
    if (v < bot[TOPK - 1]) {
      int j = TOPK - 1;
      while (j > 0 && bot[j - 1] > v) { bot[j] = bot[j - 1]; --j; }
      bot[j] = v;
    }
  }
#pragma unroll
  for (int i = 0; i < TOPK; ++i) { st[tid * TOPK + i] = top[i]; sb[tid * TOPK + i] = bot[i]; }
  __syncthreads();

  if (tid == 0) {
    float T[TOPK], B[TOPK];
#pragma unroll
    for (int i = 0; i < TOPK; ++i) { T[i] = -1e30f; B[i] = 1e30f; }
    for (int i = 0; i < 256 * TOPK; ++i) {
      float v = st[i];
      if (v > T[TOPK - 1]) {
        int j = TOPK - 1;
        while (j > 0 && T[j - 1] < v) { T[j] = T[j - 1]; --j; }
        T[j] = v;
      }
      v = sb[i];
      if (v < B[TOPK - 1]) {
        int j = TOPK - 1;
        while (j > 0 && B[j - 1] > v) { B[j] = B[j - 1]; --j; }
        B[j] = v;
      }
    }
    float part = 0.0f;
#pragma unroll
    for (int i = 0; i < TOPK; ++i)
      part += softplus_f(-T[i]) + softplus_f(B[i]);
    atomicAdd(&out[OUT_LOSS_OFF], part / (2.0f * N_SUB * TOPK));
  }
}

// ---------------------------------------------------------------------------
// launch
// ---------------------------------------------------------------------------
extern "C" void kernel_launch(void* const* d_in, const int* in_sizes, int n_in,
                              void* d_out, int out_size, void* d_ws, size_t ws_size,
                              hipStream_t stream) {
  const float* x  = (const float*)d_in[0];   // (1,1,628,512)
  const float* tf = (const float*)d_in[1];   // (8192,3072)
  const float* cw = (const float*)d_in[2];   // (16,1,3,3)
  const float* cb = (const float*)d_in[3];   // (16,)
  const float* wa = (const float*)d_in[4];   // (1024,321536)
  const float* ba = (const float*)d_in[5];   // (1024,)
  const float* wt = (const float*)d_in[6];   // (256,3072)
  const float* bt = (const float*)d_in[7];   // (256,)

  float* out = (float*)d_out;
  float* ws  = (float*)d_ws;
  float* h   = ws + WS_H_OFF;                // 16 x 321536
  float* sc  = ws + WS_SC_OFF;               // 4 x 16 x 8192 scores -> probs

  // 0) bias-init audio accumulator + zero loss slot
  init_out_kernel<<<(16 * AUDIO_N + 256) / 256, 256, 0, stream>>>(ba, out);
  // 1) conv + relu
  conv_relu_kernel<<<(SPATIAL + 255) / 256, 256, 0, stream>>>(x, cw, cb, h);
  // 2) audio GEMM, split-K over 64 ways, 64 N-tiles, one wave each
  audio_gemm_kernel<<<64 * SPLITK, 32, 0, stream>>>(h, wa, out);
  // 3) text GEMM, 8192 tiles
  text_gemm_kernel<<<TEXT_M / 16 * (TEXT_N / 16), 32, 0, stream>>>(tf, wt, bt, out);
  // 4) scores GEMM, 4 groups x 512 tiles
  scores_gemm_kernel<<<N_SUB * (TEXT_M / 16), 32, 0, stream>>>(out, sc);
  // 5) softmax over each of 64 rows
  softmax_kernel<<<N_SUB * 16, 256, 0, stream>>>(sc);
  // 6) top/bottom-5 + loss
  topk_loss_kernel<<<N_SUB, 256, 0, stream>>>(sc, out);
}
